// Attention_2388001816935
// MI455X (gfx1250) — compile-verified, matrix operand order checked by reading, not run on hardware
//
#include <hip/hip_runtime.h>
#include <hip/hip_bf16.h>

// ---------------------------------------------------------------------------
// MHA forward for MI455X (gfx1250, wave32, WMMA).
//   B=2, L=2048, D_MODEL=1024, H=16, Dh=64.
//   Round 3: software-pipelined double-buffered LDS staging (async
//   global->LDS for the K tile, pipelined VGPR staging elsewhere), one
//   barrier per K-step, batched ds_load fragment feeds before WMMA groups,
//   L2 prefetch two tiles ahead.
// ---------------------------------------------------------------------------

typedef __attribute__((ext_vector_type(16))) _Float16 v16h;
typedef __attribute__((ext_vector_type(8)))  _Float16 v8h;
typedef __attribute__((ext_vector_type(8)))  float    v8f;

#define B_   2
#define L_   2048
#define H_   16
#define DH_  64
#define DM_  1024
#define BIG_NEG_ 1e10f

static __device__ __forceinline__ v8f zero8() {
    v8f z = {0.f, 0.f, 0.f, 0.f, 0.f, 0.f, 0.f, 0.f};
    return z;
}

// Build a 16x32 f16 A/B fragment for one lane from two contiguous 8-half runs.
// Per ISA layout: lane holds row (A) / col (B) = lane&15; elements 0..7 are
// K = koff..koff+7, elements 8..15 are K = koff+16..koff+23, koff=(lane>>4)*8.
static __device__ __forceinline__ v16h make_frag(const _Float16* lo,
                                                 const _Float16* hi) {
    v8h a = *(const v8h*)lo;
    v8h b = *(const v8h*)hi;
    return __builtin_shufflevector(a, b, 0, 1, 2, 3, 4, 5, 6, 7,
                                         8, 9, 10, 11, 12, 13, 14, 15);
}

static __device__ __forceinline__ v8f wmma16(v16h a, v16h b, v8f c) {
    return __builtin_amdgcn_wmma_f32_16x16x32_f16(
        /*neg_a=*/false, a, /*neg_b=*/false, b,
        /*c_mod=*/(short)0, c, /*reuse_a=*/false, /*reuse_b=*/false);
}

// Async copy of 16 bytes global -> LDS (tracked by ASYNCcnt, wave32 per-lane).
// lds_off = low 32 bits of the generic shared-memory address (LDS aperture
// keeps the byte offset in addr[31:0]).
static __device__ __forceinline__ void async_copy_b128(uint32_t lds_off,
                                                       const void* gptr) {
    asm volatile("global_load_async_to_lds_b128 %0, %1, off"
                 :
                 : "v"(lds_off), "v"((uint64_t)(uintptr_t)gptr)
                 : "memory");
}

static __device__ __forceinline__ void wait_asynccnt0() {
    asm volatile("s_wait_asynccnt 0x0" ::: "memory");
}

// ---------------------------------------------------------------------------
// Projection: Y[b,h,l,d] = sum_k X[b*L+l, k] * W[k, h*64+d], output f16.
// Block = 256 threads (8 waves), tile M=128, N=64 (one head), K-step 32.
// Double-buffered pipeline: load k-step i+1 globals early, compute step i,
// convert+store step i+1 into the other LDS buffer, single barrier.
// ---------------------------------------------------------------------------
__global__ __launch_bounds__(256) void proj_f16_kernel(
    const float* __restrict__ X,     // [4096, 1024]
    const float* __restrict__ W,     // [1024, 1024]
    _Float16* __restrict__ Y)        // [B, H, L, 64]
{
    __shared__ __align__(16) _Float16 As[2][128][32];  // As[buf][m][k]
    __shared__ __align__(16) _Float16 Bt[2][64][32];   // Bt[buf][n][k]

    const int tid  = threadIdx.x;
    const int lane = tid & 31;
    const int wave = tid >> 5;
    const int h    = blockIdx.x;          // n0 = 64*h
    const int n0   = h * 64;
    const int m0   = blockIdx.y * 128;
    const int ln   = lane & 15;
    const int koff = (lane >> 4) * 8;

    // per-thread staging coordinates
    const int arow = tid >> 1, aseg = tid & 1;       // A: 16 floats each
    const int bk   = tid >> 3, bseg = tid & 7;       // B: 8 floats each

    const float* aptr = &X[(size_t)(m0 + arow) * DM_ + aseg * 16];
    const float* bptr = &W[(size_t)bk * DM_ + n0 + bseg * 8];

    // ---- prologue: stage k-step 0 into buffer 0 ----
    {
        const float4* gp = (const float4*)aptr;
        float fa[16];
        *(float4*)&fa[0]  = gp[0];
        *(float4*)&fa[4]  = gp[1];
        *(float4*)&fa[8]  = gp[2];
        *(float4*)&fa[12] = gp[3];
        v8h h0, h1;
#pragma unroll
        for (int j = 0; j < 8; ++j) {
            h0[j] = (_Float16)fa[j];
            h1[j] = (_Float16)fa[8 + j];
        }
        *(v8h*)&As[0][arow][aseg * 16]     = h0;
        *(v8h*)&As[0][arow][aseg * 16 + 8] = h1;

        const float4* gw = (const float4*)bptr;
        float fb[8];
        *(float4*)&fb[0] = gw[0];
        *(float4*)&fb[4] = gw[1];
#pragma unroll
        for (int j = 0; j < 8; ++j)
            Bt[0][bseg * 8 + j][bk] = (_Float16)fb[j];
    }
    __syncthreads();

    v8f acc[4];
#pragma unroll
    for (int t = 0; t < 4; ++t) acc[t] = zero8();

    for (int kb = 0; kb < DM_; kb += 32) {
        const int cur = (kb >> 5) & 1, nxt = cur ^ 1;
        const bool has_next = (kb + 32 < DM_);

        // ---- issue global loads for k-step i+1 (latency hides behind WMMA)
        float fa[16], fb[8];
        if (has_next) {
            const float4* gp = (const float4*)(aptr + kb + 32);
            *(float4*)&fa[0]  = gp[0];
            *(float4*)&fa[4]  = gp[1];
            *(float4*)&fa[8]  = gp[2];
            *(float4*)&fa[12] = gp[3];
            const float4* gw = (const float4*)(bptr + (size_t)(kb + 32) * DM_);
            *(float4*)&fb[0] = gw[0];
            *(float4*)&fb[4] = gw[1];
        }
        if (kb + 64 < DM_) {   // L2 warm-up two steps ahead
            __builtin_prefetch(aptr + kb + 64, 0, 1);
            __builtin_prefetch(bptr + (size_t)(kb + 64) * DM_, 0, 1);
        }

        // ---- compute k-step i: batch fragment loads, then 4 WMMAs ----
        const int crow = wave * 16 + ln;
        v16h a = make_frag(&As[cur][crow][koff], &As[cur][crow][koff + 16]);
        v16h bf[4];
#pragma unroll
        for (int t = 0; t < 4; ++t) {
            const int nc = t * 16 + ln;
            bf[t] = make_frag(&Bt[cur][nc][koff], &Bt[cur][nc][koff + 16]);
        }
#pragma unroll
        for (int t = 0; t < 4; ++t) acc[t] = wmma16(a, bf[t], acc[t]);

        // ---- convert + store k-step i+1 into the other buffer ----
        if (has_next) {
            v8h h0, h1;
#pragma unroll
            for (int j = 0; j < 8; ++j) {
                h0[j] = (_Float16)fa[j];
                h1[j] = (_Float16)fa[8 + j];
            }
            *(v8h*)&As[nxt][arow][aseg * 16]     = h0;
            *(v8h*)&As[nxt][arow][aseg * 16 + 8] = h1;
#pragma unroll
            for (int j = 0; j < 8; ++j)
                Bt[nxt][bseg * 8 + j][bk] = (_Float16)fb[j];
        }
        __syncthreads();
    }

    // ---- write C tiles as f16 in [B,H,L,64] layout ----
#pragma unroll
    for (int t = 0; t < 4; ++t) {
#pragma unroll
        for (int r = 0; r < 8; ++r) {
            const int mrow = m0 + wave * 16 + r + ((lane >> 4) * 8);
            const int b = mrow >> 11;            // /L_
            const int l = mrow & (L_ - 1);
            const int d = t * 16 + ln;
            Y[(((size_t)b * H_ + h) * L_ + l) * DH_ + d] = (_Float16)acc[t][r];
        }
    }
}

// ---------------------------------------------------------------------------
// Flash attention: per block 128 q rows (16 per wave, 8 waves) of one (b,h).
// Key loop in steps of 32; double-buffered LDS: K tile staged with async
// global->LDS into the next buffer while WMMAs run on the current one.
// ---------------------------------------------------------------------------
__global__ __launch_bounds__(256) void attn_kernel(
    const _Float16* __restrict__ QW,   // [B,H,L,64]
    const _Float16* __restrict__ KW,   // [B,H,L,64]
    const _Float16* __restrict__ VW,   // [B,H,L,64]
    const float* __restrict__ v_mask,  // [B,L]
    const float* __restrict__ q_mask,  // [B,L]
    float* __restrict__ O)             // [B,L,1024]
{
    __shared__ __align__(16) _Float16 Ks[2][32 * 64];  // Ks[buf][k*64+d]
    __shared__ __align__(16) _Float16 Vt[2][64][32];   // Vt[buf][d][k]
    __shared__ __align__(16) _Float16 Ps[8][16][32];   // per-wave P tile

    const int tid  = threadIdx.x;
    const int lane = tid & 31;
    const int wave = tid >> 5;
    const int h = blockIdx.y, b = blockIdx.z;
    const size_t bh = (size_t)b * H_ + h;
    const int q0   = blockIdx.x * 128 + wave * 16;
    const int ln   = lane & 15;
    const int koff = (lane >> 4) * 8;

    // Q fragments for both 32-wide halves of the head dim; reused all loop.
    const _Float16* qbase = QW + (bh * L_ + (q0 + ln)) * DH_;
    const v16h qa0 = make_frag(qbase + koff,      qbase + koff + 16);
    const v16h qa1 = make_frag(qbase + 32 + koff, qbase + 32 + koff + 16);

    // staging coordinates
    const uint32_t ks_off[2] = {
        (uint32_t)(uintptr_t)(&Ks[0][0]) + (uint32_t)tid * 16u,
        (uint32_t)(uintptr_t)(&Ks[1][0]) + (uint32_t)tid * 16u};
    const int vk = tid >> 3, vseg = tid & 7;          // V: 8 halves each
    const _Float16* kbase = KW + (bh * L_) * DH_ + tid * 8;
    const _Float16* vbase = VW + (bh * L_ + vk) * DH_ + vseg * 8;

    // ---- prologue: stage key-step 0 into buffer 0 ----
    async_copy_b128(ks_off[0], kbase);
    {
        v8h v0 = *(const v8h*)vbase;
#pragma unroll
        for (int j = 0; j < 8; ++j) Vt[0][vseg * 8 + j][vk] = v0[j];
    }
    wait_asynccnt0();
    __syncthreads();

    v8f o[4];
#pragma unroll
    for (int t = 0; t < 4; ++t) o[t] = zero8();
    float mi[8], li[8];
#pragma unroll
    for (int r = 0; r < 8; ++r) { mi[r] = -1e30f; li[r] = 0.f; }

    for (int k0 = 0; k0 < L_; k0 += 32) {
        const int cur = (k0 >> 5) & 1, nxt = cur ^ 1;
        const bool has_next = (k0 + 32 < L_);

        // ---- stage key-step i+1: async K into other buffer, V into VGPRs
        v8h vnext;
        if (has_next) {
            async_copy_b128(ks_off[nxt], kbase + (size_t)(k0 + 32) * DH_);
            vnext = *(const v8h*)(vbase + (size_t)(k0 + 32) * DH_);
        }
        if (k0 + 64 < L_) {   // L2 warm-up two steps ahead
            __builtin_prefetch(kbase + (size_t)(k0 + 64) * DH_, 0, 1);
            __builtin_prefetch(vbase + (size_t)(k0 + 64) * DH_, 0, 1);
        }

        // ---- S = Q K^T (batch all K fragments, then 4 WMMAs) ----
        const _Float16* kcur = &Ks[cur][0];
        v16h kf0[2], kf1[2];
#pragma unroll
        for (int t = 0; t < 2; ++t) {
            const int kc = t * 16 + ln;
            kf0[t] = make_frag(kcur + kc * 64 + koff,
                               kcur + kc * 64 + koff + 16);
            kf1[t] = make_frag(kcur + kc * 64 + 32 + koff,
                               kcur + kc * 64 + 32 + koff + 16);
        }
        v8f s[2];
#pragma unroll
        for (int t = 0; t < 2; ++t) {
            v8f z = zero8();
            z = wmma16(qa0, kf0[t], z);
            z = wmma16(qa1, kf1[t], z);
            s[t] = z;
        }

        // additive key mask (column = this lane's key)
        float pen0 = (1.0f - v_mask[(size_t)b * L_ + k0 + ln])      * BIG_NEG_;
        float pen1 = (1.0f - v_mask[(size_t)b * L_ + k0 + 16 + ln]) * BIG_NEG_;

        // ---- online softmax (per C-layout row; 16-lane reductions) ----
        float p0[8], p1[8], rescale[8];
#pragma unroll
        for (int r = 0; r < 8; ++r) {
            float a0 = s[0][r] * 0.125f - pen0;   // 1/sqrt(64)
            float a1 = s[1][r] * 0.125f - pen1;
            float mx = fmaxf(a0, a1);
            mx = fmaxf(mx, __shfl_xor(mx, 1));
            mx = fmaxf(mx, __shfl_xor(mx, 2));
            mx = fmaxf(mx, __shfl_xor(mx, 4));
            mx = fmaxf(mx, __shfl_xor(mx, 8));
            float mnew  = fmaxf(mi[r], mx);
            float scale = __expf(mi[r] - mnew);
            float e0 = __expf(a0 - mnew);
            float e1 = __expf(a1 - mnew);
            float sm = e0 + e1;
            sm += __shfl_xor(sm, 1);
            sm += __shfl_xor(sm, 2);
            sm += __shfl_xor(sm, 4);
            sm += __shfl_xor(sm, 8);
            li[r] = li[r] * scale + sm;
            mi[r] = mnew;
            p0[r] = e0;
            p1[r] = e1;
            rescale[r] = scale;
        }
#pragma unroll
        for (int t = 0; t < 4; ++t)
#pragma unroll
            for (int r = 0; r < 8; ++r)
                o[t][r] = o[t][r] * rescale[r];

        // ---- re-layout P (C layout -> A fragment) via per-wave LDS ----
        const int rhi = (lane >> 4) * 8;
#pragma unroll
        for (int r = 0; r < 8; ++r) {
            Ps[wave][rhi + r][ln]      = (_Float16)p0[r];
            Ps[wave][rhi + r][16 + ln] = (_Float16)p1[r];
        }
        v16h pa = make_frag(&Ps[wave][ln][koff], &Ps[wave][ln][koff + 16]);

        // ---- O += P * V (batch all V fragments, then 4 WMMAs) ----
        v16h vb[4];
#pragma unroll
        for (int t = 0; t < 4; ++t) {
            const int dc = t * 16 + ln;
            vb[t] = make_frag(&Vt[cur][dc][koff], &Vt[cur][dc][koff + 16]);
        }
#pragma unroll
        for (int t = 0; t < 4; ++t) o[t] = wmma16(pa, vb[t], o[t]);

        // ---- scatter next V tile into the other buffer, then barrier ----
        if (has_next) {
#pragma unroll
            for (int j = 0; j < 8; ++j) Vt[nxt][vseg * 8 + j][vk] = vnext[j];
        }
        wait_asynccnt0();
        __syncthreads();
    }

    // ---- epilogue: normalize, apply q_mask, store f32 ----
#pragma unroll
    for (int r = 0; r < 8; ++r) {
        const int q = q0 + r + ((lane >> 4) * 8);
        const float inv = 1.0f / fmaxf(li[r], 1e-20f);
        const float qm  = q_mask[(size_t)b * L_ + q];
#pragma unroll
        for (int t = 0; t < 4; ++t) {
            O[((size_t)b * L_ + q) * (H_ * DH_) + h * DH_ + t * 16 + ln] =
                o[t][r] * inv * qm;
        }
    }
}

// ---------------------------------------------------------------------------
extern "C" void kernel_launch(void* const* d_in, const int* in_sizes, int n_in,
                              void* d_out, int out_size, void* d_ws,
                              size_t ws_size, hipStream_t stream) {
    (void)in_sizes; (void)n_in; (void)out_size; (void)ws_size;
    const float* q      = (const float*)d_in[0];
    const float* k      = (const float*)d_in[1];
    const float* v      = (const float*)d_in[2];
    const float* v_mask = (const float*)d_in[3];
    const float* q_mask = (const float*)d_in[4];
    const float* q_kern = (const float*)d_in[5];
    const float* k_kern = (const float*)d_in[6];
    const float* v_kern = (const float*)d_in[7];
    float* out = (float*)d_out;

    const size_t per = (size_t)B_ * H_ * L_ * DH_;   // 4,194,304 halves = 8 MB
    _Float16* qw = (_Float16*)d_ws;
    _Float16* kw = qw + per;
    _Float16* vw = kw + per;

    dim3 pgrid(DM_ / 64, (B_ * L_) / 128);           // (16, 32)
    proj_f16_kernel<<<pgrid, 256, 0, stream>>>(q, q_kern, qw);
    proj_f16_kernel<<<pgrid, 256, 0, stream>>>(k, k_kern, kw);
    proj_f16_kernel<<<pgrid, 256, 0, stream>>>(v, v_kern, vw);

    dim3 agrid(L_ / 128, H_, B_);                    // (16, 16, 2)
    attn_kernel<<<agrid, 256, 0, stream>>>(qw, kw, vw, v_mask, q_mask, out);
}
